// DeepCache_18708877541862
// MI455X (gfx1250) — matrix operand order repaired and checked by reading, not compiled, not run on hardware
//
#include <hip/hip_runtime.h>
#include <cstddef>

typedef __attribute__((ext_vector_type(16))) _Float16 v16h;
typedef __attribute__((ext_vector_type(8)))  float    v8f;

#define B_  64
#define S_  2048
#define EMB_ 32
#define IN_  64
#define H_   128
#define OUT_ 256

__device__ __forceinline__ float sigm(float x) { return 1.0f / (1.0f + expf(-x)); }

// A-matrix fragment (16-bit, 16x32 tile) in contiguous k order (same as B):
// lane = (kh<<4)|m ; element e -> K = k0 + kh*16 + e
__device__ __forceinline__ v16h load_a_frag(const _Float16* __restrict__ row, int kh, int k0) {
  v16h a;
#pragma unroll
  for (int e = 0; e < 16; ++e) {
    int k = k0 + kh * 16 + e;
    a[e] = row[k];
  }
  return a;
}

// ---------------------------------------------------------------------------
// Kernel 1: byte decomposition + embedding gather + encoder MLP -> emb f16
// emb layout: [B][S][64] (first 32 = pc_e, last 32 = ad_e)
// ---------------------------------------------------------------------------
__global__ void embed_encode_kernel(const int* __restrict__ inp,
                                    const float* __restrict__ pc_emb,
                                    const float* __restrict__ addr_emb,
                                    const float* __restrict__ enc_W,
                                    const float* __restrict__ enc_b,
                                    _Float16* __restrict__ emb) {
  int idx = blockIdx.x * blockDim.x + threadIdx.x;
  if (idx >= B_ * S_ * 2) return;
  int which = idx & 1;
  int pos = idx >> 1;                       // b*S + s
  int val = inp[(size_t)pos * 2 + which];
  const float* tabs = which ? addr_emb : pc_emb;
  const float* trow[4];
#pragma unroll
  for (int i = 0; i < 4; ++i) {
    int by = (val >> (24 - 8 * i)) & 0xFF;
    trow[i] = tabs + ((size_t)i * 256 + by) * EMB_;
  }
  _Float16* out = emb + (size_t)pos * 64 + which * EMB_;
  for (int j = 0; j < EMB_; ++j) {
    float acc = enc_b[j];
    const float* wrow = enc_W + j * 128;
#pragma unroll
    for (int i = 0; i < 4; ++i) {
#pragma unroll 8
      for (int e = 0; e < 32; ++e) acc += trow[i][e] * wrow[i * 32 + e];
    }
    out[j] = (_Float16)sigm(acc);
  }
}

// ---------------------------------------------------------------------------
// Kernel 2: dist = mean over S of emb  -> [B][64] f32
// ---------------------------------------------------------------------------
__global__ void dist_kernel(const _Float16* __restrict__ emb, float* __restrict__ dist) {
  int idx = blockIdx.x * blockDim.x + threadIdx.x;
  if (idx >= B_ * 64) return;
  int b = idx >> 6, d = idx & 63;
  const _Float16* p = emb + (size_t)b * S_ * 64 + d;
  float acc = 0.0f;
  for (int s = 0; s < S_; ++s) acc += (float)p[(size_t)s * 64];
  dist[idx] = acc * (1.0f / (float)S_);
}

// ---------------------------------------------------------------------------
// Kernel 3: LSTM. 4 workgroups x 512 threads (16 waves); each WG owns 16 batch
// rows and runs the full 2048-step recurrence with weights resident in LDS
// (192 KB f16, pre-swizzled into WMMA B-fragment order; needs CDNA5 320KB LDS).
// ---------------------------------------------------------------------------
__global__ void __launch_bounds__(512)
lstm_kernel(const _Float16* __restrict__ emb,
            const float* __restrict__ h0, const float* __restrict__ c0,
            const float* __restrict__ W_ih, const float* __restrict__ W_hh,
            const float* __restrict__ b_ih, const float* __restrict__ b_hh,
            _Float16* __restrict__ hn_out) {
  extern __shared__ char smem[];
  _Float16* wlds   = (_Float16*)smem;                         // 32*6*32*16 halves = 192KB
  float*    gates  = (float*)(smem + 98304 * 2);              // 16*512 f32 = 32KB
  float*    c_lds  = (float*)(smem + 98304 * 2 + 32768);      // 16*128 f32 = 8KB
  float*    bias_l = (float*)(smem + 98304 * 2 + 32768 + 8192);   // 512 f32 = 2KB
  _Float16* h_lds  = (_Float16*)(smem + 98304 * 2 + 32768 + 8192 + 2048); // 16*128 f16 = 4KB

  const int tid  = threadIdx.x;
  const int lane = tid & 31;
  const int wave = tid >> 5;        // 0..15
  const int m    = lane & 15;
  const int kh   = lane >> 4;
  const int b0   = blockIdx.x * 16; // batch rows for this WG

  // Swizzle combined weights Wc[K=192][N=512] into B-fragment-contiguous LDS.
  // idx = ((nt*6 + kc)*32 + l)*16 + e ; value = Wc[kc*32 + (l>>4)*16 + e][nt*16 + (l&15)]
  for (int idx = tid; idx < 32 * 6 * 32 * 16; idx += 512) {
    int e  = idx & 15;
    int l  = (idx >> 4) & 31;
    int kc = (idx >> 9) % 6;
    int nt = (idx >> 9) / 6;
    int K  = kc * 32 + (l >> 4) * 16 + e;
    int N  = nt * 16 + (l & 15);
    float w = (K < 64) ? W_ih[(size_t)N * 64 + K] : W_hh[(size_t)N * 128 + (K - 64)];
    wlds[idx] = (_Float16)w;
  }
  for (int idx = tid; idx < 16 * 128; idx += 512) {
    h_lds[idx] = (_Float16)h0[(size_t)b0 * 128 + idx];
    c_lds[idx] = c0[(size_t)b0 * 128 + idx];
  }
  bias_l[tid] = b_ih[tid] + b_hh[tid];
  __syncthreads();

  const int nt0 = wave * 2;
  const int n   = lane & 15;

  for (int t = 0; t < S_; ++t) {
    const _Float16* xrow = emb + ((size_t)(b0 + m) * S_ + t) * 64;
    __builtin_prefetch(xrow + 64, 0, 0);  // next timestep -> global_prefetch_b8
    v16h xf0 = load_a_frag(xrow, kh, 0);
    v16h xf1 = load_a_frag(xrow, kh, 32);
    const _Float16* hrow = h_lds + m * 128;
    v16h hf0 = load_a_frag(hrow, kh, 0);
    v16h hf1 = load_a_frag(hrow, kh, 32);
    v16h hf2 = load_a_frag(hrow, kh, 64);
    v16h hf3 = load_a_frag(hrow, kh, 96);

#pragma unroll
    for (int j = 0; j < 2; ++j) {
      const int nt = nt0 + j;
      const int nb = nt * 16;
      float bv = bias_l[nb + n];
      v8f c = {bv, bv, bv, bv, bv, bv, bv, bv};
      const _Float16* wp = wlds + (size_t)nt * 6 * 32 * 16;
#pragma unroll
      for (int kc = 0; kc < 6; ++kc) {
        const _Float16* p = wp + (kc * 32 + lane) * 16;
        v16h bf;
#pragma unroll
        for (int e = 0; e < 16; ++e) bf[e] = p[e];
        v16h af = (kc == 0) ? xf0 : (kc == 1) ? xf1 : (kc == 2) ? hf0
                : (kc == 3) ? hf1 : (kc == 4) ? hf2 : hf3;
        c = __builtin_amdgcn_wmma_f32_16x16x32_f16(false, af, false, bf,
                                                   (short)0, c, false, false);
      }
#pragma unroll
      for (int r = 0; r < 8; ++r) {
        int row = (lane < 16) ? r : (r + 8);
        gates[row * 512 + nb + n] = c[r];
      }
    }
    __syncthreads();

    for (int idx = tid; idx < 16 * 128; idx += 512) {
      int mm = idx >> 7, d = idx & 127;
      const float* g = gates + mm * 512;
      float gi = g[d], gf = g[128 + d], gg = g[256 + d], go = g[384 + d];
      float cc = sigm(gf) * c_lds[idx] + sigm(gi) * tanhf(gg);
      c_lds[idx] = cc;
      h_lds[idx] = (_Float16)(sigm(go) * tanhf(cc));
    }
    __syncthreads();
  }

  for (int idx = tid; idx < 16 * 128; idx += 512)
    hn_out[(size_t)b0 * 128 + idx] = h_lds[idx];
}

// ---------------------------------------------------------------------------
// Kernel 4: decoder logits[k][b][o] = h_n @ dec_W[k]^T + dec_b[k]  (WMMA)
// grid = 16 blocks (k*4+mtile), 512 threads = 16 waves = 16 N-tiles.
// ---------------------------------------------------------------------------
__global__ void __launch_bounds__(512)
decoder_kernel(const _Float16* __restrict__ hn,
               const float* __restrict__ dec_W, const float* __restrict__ dec_b,
               float* __restrict__ logits) {
  const int k     = blockIdx.x >> 2;
  const int mtile = blockIdx.x & 3;
  const int lane  = threadIdx.x & 31;
  const int wave  = threadIdx.x >> 5;   // ntile 0..15
  const int m = lane & 15, kh = lane >> 4, n = lane & 15;
  const int n0 = wave * 16;

  const _Float16* arow = hn + (size_t)(mtile * 16 + m) * 128;
  v16h a0 = load_a_frag(arow, kh, 0);
  v16h a1 = load_a_frag(arow, kh, 32);
  v16h a2 = load_a_frag(arow, kh, 64);
  v16h a3 = load_a_frag(arow, kh, 96);

  float bv = dec_b[k * 256 + n0 + n];
  v8f c = {bv, bv, bv, bv, bv, bv, bv, bv};
#pragma unroll
  for (int kc = 0; kc < 4; ++kc) {
    // B[kk][nn] = dec_W[k][nn][kk]; kk = kc*32 + kh*16 + e  (contiguous in e)
    const float* bp = dec_W + ((size_t)k * 256 + n0 + n) * 128 + kc * 32 + kh * 16;
    v16h bf;
#pragma unroll
    for (int e = 0; e < 16; ++e) bf[e] = (_Float16)bp[e];
    v16h af = (kc == 0) ? a0 : (kc == 1) ? a1 : (kc == 2) ? a2 : a3;
    c = __builtin_amdgcn_wmma_f32_16x16x32_f16(false, af, false, bf,
                                               (short)0, c, false, false);
  }
#pragma unroll
  for (int r = 0; r < 8; ++r) {
    int row = mtile * 16 + ((lane < 16) ? r : (r + 8));
    logits[((size_t)k * 64 + row) * 256 + n0 + n] = c[r];
  }
}

// ---------------------------------------------------------------------------
// Kernel 5: low-temperature softmax + be[b][k][:] = probs @ addr_emb[k]
// one block per (k,b), 256 threads.
// ---------------------------------------------------------------------------
__global__ void softmax_be_kernel(const float* __restrict__ logits,
                                  const float* __restrict__ addr_emb,
                                  float* __restrict__ probs,
                                  float* __restrict__ be) {
  __shared__ float sh[256];
  __shared__ float pl[256];
  const int k = blockIdx.x >> 6, b = blockIdx.x & 63;
  const int t = threadIdx.x;
  float l = logits[((size_t)k * 64 + b) * 256 + t];
  sh[t] = l;
  __syncthreads();
  for (int s = 128; s > 0; s >>= 1) {
    if (t < s) sh[t] = fmaxf(sh[t], sh[t + s]);
    __syncthreads();
  }
  float lmax = sh[0];
  __syncthreads();
  float e = expf((l - lmax) * 1000.0f);   // / TEMP, TEMP = 1e-3
  sh[t] = e;
  __syncthreads();
  for (int s = 128; s > 0; s >>= 1) {
    if (t < s) sh[t] += sh[t + s];
    __syncthreads();
  }
  float pv = e / sh[0];
  probs[((size_t)k * 64 + b) * 256 + t] = pv;
  pl[t] = pv;
  __syncthreads();
  if (t < 32) {
    float acc = 0.0f;
    for (int o = 0; o < 256; ++o) acc += pl[o] * addr_emb[((size_t)k * 256 + o) * 32 + t];
    be[((size_t)b * 4 + k) * 32 + t] = acc;   // [b][k*32+e] == reshape(b, 128)
  }
}

// ---------------------------------------------------------------------------
// Kernel 6: fe = sigmoid(enc(be)); fr = sigmoid(relu([fe,dist]@fc1^T)@fc2^T)
// ---------------------------------------------------------------------------
__global__ void fr_kernel(const float* __restrict__ be, const float* __restrict__ dist,
                          const float* __restrict__ enc_W, const float* __restrict__ enc_b,
                          const float* __restrict__ fc1_W, const float* __restrict__ fc1_b,
                          const float* __restrict__ fc2_W, const float* __restrict__ fc2_b,
                          float* __restrict__ fr0, float* __restrict__ fr1) {
  int b = threadIdx.x;
  if (b >= B_) return;
  float fin[96];
  for (int j = 0; j < 32; ++j) {
    float acc = enc_b[j];
    const float* wrow = enc_W + j * 128;
    for (int i = 0; i < 128; ++i) acc += be[b * 128 + i] * wrow[i];
    fin[j] = sigm(acc);
  }
  for (int d = 0; d < 64; ++d) fin[32 + d] = dist[b * 64 + d];
  float r1[10];
  for (int u = 0; u < 10; ++u) {
    float acc = fc1_b[u];
    for (int i = 0; i < 96; ++i) acc += fin[i] * fc1_W[u * 96 + i];
    r1[u] = fmaxf(acc, 0.0f);
  }
  float o0 = fc2_b[0], o1 = fc2_b[1];
  for (int u = 0; u < 10; ++u) { o0 += r1[u] * fc2_W[u]; o1 += r1[u] * fc2_W[10 + u]; }
  fr0[b] = sigm(o0);
  fr1[b] = sigm(o1);
}

// ---------------------------------------------------------------------------
extern "C" void kernel_launch(void* const* d_in, const int* in_sizes, int n_in,
                              void* d_out, int out_size, void* d_ws, size_t ws_size,
                              hipStream_t stream) {
  (void)in_sizes; (void)n_in; (void)out_size; (void)ws_size;
  const int*   inp      = (const int*)d_in[0];
  const float* h0       = (const float*)d_in[1];
  const float* c0       = (const float*)d_in[2];
  const float* pc_emb   = (const float*)d_in[3];
  const float* addr_emb = (const float*)d_in[4];
  const float* enc_W    = (const float*)d_in[5];
  const float* enc_b    = (const float*)d_in[6];
  const float* W_ih     = (const float*)d_in[7];
  const float* W_hh     = (const float*)d_in[8];
  const float* b_ih     = (const float*)d_in[9];
  const float* b_hh     = (const float*)d_in[10];
  const float* dec_W    = (const float*)d_in[11];
  const float* dec_b    = (const float*)d_in[12];
  const float* fc1_W    = (const float*)d_in[13];
  const float* fc1_b    = (const float*)d_in[14];
  const float* fc2_W    = (const float*)d_in[15];
  const float* fc2_b    = (const float*)d_in[16];

  char* ws = (char*)d_ws;
  size_t off = 0;
  _Float16* emb = (_Float16*)(ws + off); off += (size_t)B_ * S_ * 64 * 2;   // 16 MB
  _Float16* hn  = (_Float16*)(ws + off); off += (size_t)B_ * H_ * 2;
  off = (off + 255) & ~(size_t)255;
  float* dist = (float*)(ws + off); off += (size_t)B_ * 64 * 4;
  float* be   = (float*)(ws + off); off += (size_t)B_ * 128 * 4;

  float* outp   = (float*)d_out;
  float* probs  = outp;                 // [4][1][64][256]
  float* logits = outp + 65536;         // [4][1][64][256]
  float* fr0    = outp + 131072;        // [64]
  float* fr1    = outp + 131136;        // [64]

  embed_encode_kernel<<<(B_ * S_ * 2 + 255) / 256, 256, 0, stream>>>(
      inp, pc_emb, addr_emb, enc_W, enc_b, emb);
  dist_kernel<<<(B_ * 64 + 255) / 256, 256, 0, stream>>>(emb, dist);

  // LDS: 192KB weights + 32KB gates + 8KB c + 2KB bias + 4KB h = ~238KB (CDNA5: 320KB/WGP)
  size_t lds_bytes = (size_t)98304 * 2 + 32768 + 8192 + 2048 + 4096;
  lstm_kernel<<<4, 512, lds_bytes, stream>>>(emb, h0, c0, W_ih, W_hh, b_ih, b_hh, hn);

  decoder_kernel<<<16, 512, 0, stream>>>(hn, dec_W, dec_b, logits);
  softmax_be_kernel<<<256, 256, 0, stream>>>(logits, addr_emb, probs, be);
  fr_kernel<<<1, 64, 0, stream>>>(be, dist, enc_W, enc_b, fc1_W, fc1_b, fc2_W, fc2_b,
                                  fr0, fr1);
}
